// DecoderBlock_22771916603726
// MI455X (gfx1250) — matrix operand order repaired and checked by reading, not compiled, not run on hardware
//
#include <hip/hip_runtime.h>

#define DMODEL 1024
#define DFFN   4096
#define SEQ    2048
#define NBATCH 4
#define NEG_INF (-__builtin_inff())

typedef __attribute__((ext_vector_type(8)))  __bf16 v8bf;
typedef __attribute__((ext_vector_type(16))) __bf16 v16bf;
typedef __attribute__((ext_vector_type(8)))  float  v8f;

// ---------------------------------------------------------------------------
// CDNA5 async global->LDS copy path (ASYNCcnt), gated so compilation cannot
// regress if this toolchain doesn't expose the builtins.
// Builtin signature (from hipcc diagnostic): param0 = int __vector(4) AS1*,
// param1 = LDS pointer, then imm offset, imm cpol.
// ---------------------------------------------------------------------------
#if defined(__gfx1250__) &&                                                    \
    __has_builtin(__builtin_amdgcn_global_load_async_to_lds_b128) &&           \
    __has_builtin(__builtin_amdgcn_s_wait_asynccnt)
#define USE_ASYNC_LDS 1
#else
#define USE_ASYNC_LDS 0
#endif

typedef int gcc_v4i __attribute__((vector_size(16)));
typedef __attribute__((address_space(1))) gcc_v4i* as1_v4i_ptr;
typedef __attribute__((address_space(3))) gcc_v4i* as3_v4i_ptr;

__device__ __forceinline__ void copy16_g2s(void* dst_lds, const void* src_global) {
#if USE_ASYNC_LDS
  __builtin_amdgcn_global_load_async_to_lds_b128(
      (as1_v4i_ptr)(size_t)src_global,   // cast away const, generic -> AS1
      (as3_v4i_ptr)dst_lds, 0, 0);
#else
  *(v8bf*)dst_lds = *(const v8bf*)src_global;
#endif
}

__device__ __forceinline__ void wait_async_le4() {
#if USE_ASYNC_LDS
  __builtin_amdgcn_s_wait_asynccnt(4);
#endif
}
__device__ __forceinline__ void wait_async_0() {
#if USE_ASYNC_LDS
  __builtin_amdgcn_s_wait_asynccnt(0);
#endif
}

// ---------------------------------------------------------------------------
// WMMA helpers (CDNA5 16x16x32 bf16, f32 accumulate)
// ---------------------------------------------------------------------------
__device__ __forceinline__ v8f wmma_bf16(v16bf a, v16bf b, v8f c) {
  // 8 args: (neg_a, A, neg_b, B, c_mod, C, reuse_a, reuse_b)
  return __builtin_amdgcn_wmma_f32_16x16x32_bf16(false, a, false, b, (short)0, c,
                                                 false, false);
}

// A-matrix 16x32 bf16 fragment from row-major storage (ld in elements).
// ISA layout: lanes 0-15 -> M=lane, K in {0..7} U {16..23};
//             lanes 16-31 -> M=lane-16, K in {8..15} U {24..31}.
__device__ __forceinline__ v16bf load_a16(const __bf16* base, int ld) {
  const int lane = (int)(threadIdx.x & 31u);
  const __bf16* p = base + (size_t)(lane & 15) * ld + ((lane & 16) ? 8 : 0);
  v8bf lo = *(const v8bf*)(p);
  v8bf hi = *(const v8bf*)(p + 16);
  v16bf a;
#pragma unroll
  for (int i = 0; i < 8; ++i) { a[i] = lo[i]; a[i + 8] = hi[i]; }
  return a;
}

// B-matrix 32x16 fragment where memory holds B^T row-major: row n = output
// column, contiguous K. ISA layout: lanes 0-15 -> N=lane, K in {0..7} U {16..23};
// lanes 16-31 -> N=lane-16, K in {8..15} U {24..31}.
__device__ __forceinline__ v16bf load_bT16(const __bf16* base, int ld) {
  const int lane = (int)(threadIdx.x & 31u);
  const __bf16* p = base + (size_t)(lane & 15) * ld + ((lane & 16) ? 8 : 0);
  v8bf lo = *(const v8bf*)(p);
  v8bf hi = *(const v8bf*)(p + 16);
  v16bf b;
#pragma unroll
  for (int i = 0; i < 8; ++i) { b[i] = lo[i]; b[i + 8] = hi[i]; }
  return b;
}

// ---------------------------------------------------------------------------
// Converters
// ---------------------------------------------------------------------------
__global__ __launch_bounds__(256) void to_bf16_kernel(const float* __restrict__ in,
                                                      __bf16* __restrict__ out,
                                                      size_t n) {
  size_t i = (size_t)blockIdx.x * blockDim.x + threadIdx.x;
  size_t stride = (size_t)gridDim.x * blockDim.x;
  for (; i < n; i += stride) out[i] = (__bf16)in[i];
}

// in[R][C] (f32) -> out[C][R] (bf16)
__global__ __launch_bounds__(256) void transpose_bf16_kernel(const float* __restrict__ in,
                                                             __bf16* __restrict__ out,
                                                             int R, int C) {
  size_t n = (size_t)R * C;
  size_t i = (size_t)blockIdx.x * blockDim.x + threadIdx.x;
  size_t stride = (size_t)gridDim.x * blockDim.x;
  for (; i < n; i += stride) {
    int r = (int)(i / C), c = (int)(i % C);
    out[(size_t)c * R + r] = (__bf16)in[i];
  }
}

// ---------------------------------------------------------------------------
// Flash attention (single head, d=1024). 8 waves/block; each wave owns a
// 128-wide slice of D. 16 query rows per block, 32 keys per iteration.
// ---------------------------------------------------------------------------
__global__ __launch_bounds__(256) void attn_wmma(const __bf16* __restrict__ Q,
                                                 const __bf16* __restrict__ KV,
                                                 float* __restrict__ out,
                                                 int causal) {
  __shared__ __attribute__((aligned(16))) __bf16 Vt[DMODEL * 32];  // V^T staged
  __shared__ __attribute__((aligned(16))) float  Sred[16 * 32];
  __shared__ __attribute__((aligned(16))) __bf16 Pls[16 * 32];
  __shared__ float mrow[16], lrow[16], crow[16];

  const int tid = threadIdx.x, wave = tid >> 5, lane = tid & 31;
  const int b = blockIdx.y, q0 = blockIdx.x * 16;
  const int cbase = wave * 128;
  const __bf16* Qb = Q + (size_t)b * SEQ * DMODEL;
  const __bf16* Kb = KV + (size_t)b * SEQ * DMODEL;
  float* Ob = out + (size_t)b * SEQ * DMODEL;

  // Q fragments for this wave's 128-wide k-slice stay resident in VGPRs.
  v16bf qf[4];
#pragma unroll
  for (int kc = 0; kc < 4; ++kc)
    qf[kc] = load_a16(Qb + (size_t)q0 * DMODEL + cbase + kc * 32, DMODEL);

  v8f acc[8] = {};
  if (tid < 16) { mrow[tid] = NEG_INF; lrow[tid] = 0.f; }

  const int kend = causal ? (q0 + 16) : SEQ;
  for (int kv = 0; kv < kend; kv += 32) {
    for (int i = tid; i < 512; i += 256) Sred[i] = 0.f;
    // L2 prefetch of the next key block (global_prefetch_b8 on gfx1250)
    {
      const int prow = kv + 32 + (tid & 31);
      if (prow < SEQ)
        __builtin_prefetch(Kb + (size_t)prow * DMODEL + (tid >> 5) * 128, 0, 1);
    }
    __syncthreads();

    // --- S partial = Q_slice * K_slice^T  (two 16x16 C-tiles, K=128 slice)
    v8f s0 = {}, s1 = {};
#pragma unroll
    for (int kc = 0; kc < 4; ++kc) {
      v16bf b0 = load_bT16(Kb + (size_t)kv * DMODEL + cbase + kc * 32, DMODEL);
      v16bf b1 = load_bT16(Kb + (size_t)(kv + 16) * DMODEL + cbase + kc * 32, DMODEL);
      s0 = wmma_bf16(qf[kc], b0, s0);
      s1 = wmma_bf16(qf[kc], b1, s1);
    }
    {
      const int col = lane & 15, half = lane >> 4;
#pragma unroll
      for (int v = 0; v < 8; ++v) {
        atomicAdd(&Sred[(v + 8 * half) * 32 + col], s0[v]);
        atomicAdd(&Sred[(v + 8 * half) * 32 + 16 + col], s1[v]);
      }
    }
    // --- stage V block transposed into LDS: Vt[c][k], c in [0,1024)
    {
      const int c0 = tid * 4;
      for (int k = 0; k < 32; ++k) {
        const __bf16* vp = Kb + (size_t)(kv + k) * DMODEL + c0;
#pragma unroll
        for (int j = 0; j < 4; ++j) Vt[(c0 + j) * 32 + k] = vp[j];
      }
    }
    __syncthreads();

    // --- online softmax over the 32-key block (one thread per query row)
    if (tid < 16) {
      const int r = tid;
      const float scale = 0.03125f;  // 1/sqrt(1024)
      float mx = mrow[r];
      float sv[32];
      float mnew = mx;
#pragma unroll
      for (int j = 0; j < 32; ++j) {
        const int key = kv + j;
        const bool ok = (key < SEQ) && (!causal || key <= q0 + r);
        sv[j] = ok ? Sred[r * 32 + j] * scale : NEG_INF;
        mnew = fmaxf(mnew, sv[j]);
      }
      const float corr = (mnew == NEG_INF) ? 1.f : __expf(mx - mnew);
      float sum = 0.f;
#pragma unroll
      for (int j = 0; j < 32; ++j) {
        const float p = (sv[j] == NEG_INF) ? 0.f : __expf(sv[j] - mnew);
        Pls[r * 32 + j] = (__bf16)p;
        sum += p;
      }
      mrow[r] = mnew;
      lrow[r] = lrow[r] * corr + sum;
      crow[r] = corr;
    }
    __syncthreads();

    // --- rescale accumulators, then acc += P(16x32) * V_slice(32x128)
    {
      const int half = lane >> 4;
      float cf[8];
#pragma unroll
      for (int v = 0; v < 8; ++v) cf[v] = crow[v + 8 * half];
      v16bf pf = load_a16(Pls, 32);
#pragma unroll
      for (int t = 0; t < 8; ++t) {
#pragma unroll
        for (int v = 0; v < 8; ++v) acc[t][v] *= cf[v];
        v16bf bv = load_bT16(&Vt[(size_t)(cbase + 16 * t) * 32], 32);
        acc[t] = wmma_bf16(pf, bv, acc[t]);
      }
    }
    __syncthreads();
  }

  // --- normalize by row sums and store f32 output
  {
    const int col = lane & 15, half = lane >> 4;
#pragma unroll
    for (int v = 0; v < 8; ++v) {
      const int r = v + 8 * half;
      const float linv = 1.f / lrow[r];
#pragma unroll
      for (int t = 0; t < 8; ++t)
        Ob[(size_t)(q0 + r) * DMODEL + cbase + 16 * t + col] = acc[t][v] * linv;
    }
  }
}

// ---------------------------------------------------------------------------
// WMMA GEMM: C[M,N] = A[M,K] (bf16, row-major) * B (given as B^T [N][K] bf16)
// + bias, optional ReLU. Writes f32 (Cf) or bf16 (Cbf).
// Block: 128x128 tile, 8 waves as 2(M) x 4(N); wave tile 64x32.
// Double-buffered LDS staging via gfx1250 async global->LDS copies
// (4 outstanding b128 ops per thread per tile; in-order completion).
// ---------------------------------------------------------------------------
__global__ __launch_bounds__(256) void gemm_wmma(const __bf16* __restrict__ A,
                                                 const __bf16* __restrict__ BT,
                                                 const float* __restrict__ bias,
                                                 float* __restrict__ Cf,
                                                 __bf16* __restrict__ Cbf,
                                                 int M, int N, int K, int relu) {
  __shared__ __attribute__((aligned(16))) __bf16 As[2][128 * 32];
  __shared__ __attribute__((aligned(16))) __bf16 Bs[2][128 * 32];
  const int tid = threadIdx.x, wave = tid >> 5, lane = tid & 31;
  const int wm = wave >> 2, wn = wave & 3;
  const int m0 = blockIdx.y * 128, n0 = blockIdx.x * 128;

  v8f acc[4][2] = {};

  // Per-thread staging footprint: 2 b128 units of As + 2 of Bs per K-tile.
  const int row0 = tid >> 2, c80 = (tid & 3) * 8;            // unit tid
  const int row1 = (tid + 256) >> 2, c81 = ((tid + 256) & 3) * 8;

#define STAGE_TILE(buf, k0)                                                    \
  do {                                                                         \
    copy16_g2s(&As[buf][row0 * 32 + c80],                                      \
               &A[(size_t)(m0 + row0) * K + (k0) + c80]);                      \
    copy16_g2s(&Bs[buf][row0 * 32 + c80],                                      \
               &BT[(size_t)(n0 + row0) * K + (k0) + c80]);                     \
    copy16_g2s(&As[buf][row1 * 32 + c81],                                      \
               &A[(size_t)(m0 + row1) * K + (k0) + c81]);                      \
    copy16_g2s(&Bs[buf][row1 * 32 + c81],                                      \
               &BT[(size_t)(n0 + row1) * K + (k0) + c81]);                     \
  } while (0)

  const int KT = K / 32;
  STAGE_TILE(0, 0);
  for (int kt = 0; kt < KT; ++kt) {
    const int cur = kt & 1;
    if (kt + 1 < KT) {
      STAGE_TILE(cur ^ 1, (kt + 1) * 32);
      wait_async_le4();   // older tile (4 ops/thread) drained; next 4 in flight
    } else {
      wait_async_0();
    }
    __syncthreads();

    v16bf b0 = load_bT16(&Bs[cur][(size_t)(wn * 32) * 32], 32);
    v16bf b1 = load_bT16(&Bs[cur][(size_t)(wn * 32 + 16) * 32], 32);
#pragma unroll
    for (int i = 0; i < 4; ++i) {
      v16bf af = load_a16(&As[cur][(size_t)(wm * 64 + 16 * i) * 32], 32);
      acc[i][0] = wmma_bf16(af, b0, acc[i][0]);
      acc[i][1] = wmma_bf16(af, b1, acc[i][1]);
    }
    __syncthreads();
  }
#undef STAGE_TILE

  const int col = lane & 15, half = lane >> 4;
#pragma unroll
  for (int i = 0; i < 4; ++i) {
#pragma unroll
    for (int j = 0; j < 2; ++j) {
      const int n = n0 + wn * 32 + 16 * j + col;
      const float bv = bias[n];
#pragma unroll
      for (int v = 0; v < 8; ++v) {
        const int m = m0 + wm * 64 + 16 * i + v + 8 * half;
        float x = acc[i][j][v] + bv;
        if (relu) x = fmaxf(x, 0.f);
        if (Cbf) Cbf[(size_t)m * N + n] = (__bf16)x;
        else     Cf[(size_t)m * N + n] = x;
      }
    }
  }
}

// ---------------------------------------------------------------------------
// Residual + LayerNorm over D=1024; one block (256 threads) per row.
// out = LN(x + r) * g + beta; optional bf16 copy for next matmul stage.
// ---------------------------------------------------------------------------
__global__ __launch_bounds__(256) void ln_residual(const float* __restrict__ x,
                                                   const float* __restrict__ r,
                                                   const float* __restrict__ g,
                                                   const float* __restrict__ be,
                                                   float* __restrict__ out,
                                                   __bf16* __restrict__ outbf) {
  __shared__ float sm[256];
  const int tid = threadIdx.x;
  const size_t row = blockIdx.x;
  const float* xr = x + row * DMODEL;
  const float* rr = r + row * DMODEL;

  float loc[4];
  float s = 0.f;
#pragma unroll
  for (int i = 0; i < 4; ++i) {
    const int c = tid * 4 + i;
    loc[i] = xr[c] + rr[c];
    s += loc[i];
  }
  sm[tid] = s;
  __syncthreads();
  for (int o = 128; o > 0; o >>= 1) {
    if (tid < o) sm[tid] += sm[tid + o];
    __syncthreads();
  }
  const float mu = sm[0] * (1.f / DMODEL);
  __syncthreads();

  float s2 = 0.f;
#pragma unroll
  for (int i = 0; i < 4; ++i) {
    const float d = loc[i] - mu;
    s2 += d * d;
  }
  sm[tid] = s2;
  __syncthreads();
  for (int o = 128; o > 0; o >>= 1) {
    if (tid < o) sm[tid] += sm[tid + o];
    __syncthreads();
  }
  const float inv = rsqrtf(sm[0] * (1.f / DMODEL) + 1e-5f);

#pragma unroll
  for (int i = 0; i < 4; ++i) {
    const int c = tid * 4 + i;
    const float o = (loc[i] - mu) * inv * g[c] + be[c];
    out[row * DMODEL + c] = o;
    if (outbf) outbf[row * DMODEL + c] = (__bf16)o;
  }
}

// ---------------------------------------------------------------------------
// Launch
// ---------------------------------------------------------------------------
extern "C" void kernel_launch(void* const* d_in, const int* in_sizes, int n_in,
                              void* d_out, int out_size, void* d_ws, size_t ws_size,
                              hipStream_t stream) {
  (void)in_sizes; (void)n_in; (void)out_size; (void)ws_size;
  const float* y   = (const float*)d_in[0];
  const float* Z   = (const float*)d_in[1];
  const float* w1  = (const float*)d_in[2];
  const float* b1  = (const float*)d_in[3];
  const float* w2  = (const float*)d_in[4];
  const float* b2  = (const float*)d_in[5];
  const float* g1  = (const float*)d_in[6];
  const float* be1 = (const float*)d_in[7];
  const float* g2  = (const float*)d_in[8];
  const float* be2 = (const float*)d_in[9];
  const float* g3  = (const float*)d_in[10];
  const float* be3 = (const float*)d_in[11];
  float* out = (float*)d_out;

  char* ws = (char*)d_ws;
  const size_t MB = 1u << 20;
  const size_t TOK = (size_t)NBATCH * SEQ;                 // 8192 rows
  __bf16* ybf  = (__bf16*)(ws);                            // 16 MB (also y1bf/y2bf)
  __bf16* Zbf  = (__bf16*)(ws + 16 * MB);                  // 16 MB
  __bf16* w1T  = (__bf16*)(ws + 32 * MB);                  // [DFFN][DMODEL], 8 MB
  __bf16* w2T  = (__bf16*)(ws + 40 * MB);                  // [DMODEL][DFFN], 8 MB
  float*  tmpA = (float*) (ws + 48 * MB);                  // 32 MB (sa/ca/ff)
  float*  tmpB = (float*) (ws + 80 * MB);                  // 32 MB (y1/y2)
  __bf16* hbf  = (__bf16*)(ws + 112 * MB);                 // [TOK][DFFN], 64 MB

  // Precision staging: f32 -> bf16 (activations), transpose+convert (weights).
  to_bf16_kernel<<<4096, 256, 0, stream>>>(y, ybf, TOK * DMODEL);
  to_bf16_kernel<<<4096, 256, 0, stream>>>(Z, Zbf, TOK * DMODEL);
  transpose_bf16_kernel<<<4096, 256, 0, stream>>>(w1, w1T, DMODEL, DFFN);
  transpose_bf16_kernel<<<4096, 256, 0, stream>>>(w2, w2T, DFFN, DMODEL);

  const dim3 agrid(SEQ / 16, NBATCH);
  // 1) causal self-attention
  attn_wmma<<<agrid, 256, 0, stream>>>(ybf, ybf, tmpA, 1);
  // 2) y1 = LN(y + sa); keep f32 in tmpB, bf16 in ybf
  ln_residual<<<(unsigned)TOK, 256, 0, stream>>>(y, tmpA, g1, be1, tmpB, ybf);
  // 3) cross-attention against Z
  attn_wmma<<<agrid, 256, 0, stream>>>(ybf, Zbf, tmpA, 0);
  // 4) y2 = LN(y1 + ca) (in-place over tmpB), bf16 in ybf
  ln_residual<<<(unsigned)TOK, 256, 0, stream>>>(tmpB, tmpA, g2, be2, tmpB, ybf);
  // 5) h = relu(y2 @ w1 + b1) -> bf16
  gemm_wmma<<<dim3(DFFN / 128, (unsigned)(TOK / 128)), 256, 0, stream>>>(
      ybf, w1T, b1, nullptr, hbf, (int)TOK, DFFN, DMODEL, 1);
  // 6) ff = h @ w2 + b2 -> f32
  gemm_wmma<<<dim3(DMODEL / 128, (unsigned)(TOK / 128)), 256, 0, stream>>>(
      hbf, w2T, b2, tmpA, nullptr, (int)TOK, DMODEL, DFFN, 0);
  // 7) y3 = LN(y2 + ff) -> d_out
  ln_residual<<<(unsigned)TOK, 256, 0, stream>>>(tmpB, tmpA, g3, be3, out, nullptr);
}
